// CNNSummarizer_2345052143924
// MI455X (gfx1250) — compile-verified, matrix operand order checked by reading, not run on hardware
//
#include <hip/hip_runtime.h>

// ---------------- WMMA types / helpers ----------------
typedef __attribute__((ext_vector_type(16))) __bf16 v16bf;
typedef __attribute__((ext_vector_type(8)))  float  v8f;
typedef __attribute__((ext_vector_type(4)))  unsigned int u32x4;

union FragB { v16bf bf; u32x4 q[2]; };

__device__ __forceinline__ unsigned short f2bf(float f) {
  unsigned int u = __float_as_uint(f);
  u += 0x7FFFu + ((u >> 16) & 1u);          // round-to-nearest-even
  return (unsigned short)(u >> 16);
}

// Two M-tiles (rows m and m+16) sharing each B fragment: halves B-side L2
// traffic and doubles WMMA density per load clause.
__device__ __forceinline__ void wmma_kloop2(const unsigned short* __restrict__ a0,
                                            const unsigned short* __restrict__ a1,
                                            const unsigned short* __restrict__ blane,
                                            int KB, int hb, v8f& acc0, v8f& acc1) {
  for (int kb = 0; kb < KB; ++kb) {
    FragB fa0, fa1, fb;
    const unsigned short* bp = blane + (size_t)kb * 512;
    fb.q[0]  = *(const u32x4*)(bp);
    fb.q[1]  = *(const u32x4*)(bp + 8);
    fa0.q[0] = *(const u32x4*)(a0 + kb * 32 + hb);
    fa0.q[1] = *(const u32x4*)(a0 + kb * 32 + 16 + hb);
    fa1.q[0] = *(const u32x4*)(a1 + kb * 32 + hb);
    fa1.q[1] = *(const u32x4*)(a1 + kb * 32 + 16 + hb);
    __builtin_prefetch(bp + 512, 0, 0);     // next K-block of B (global_prefetch_b8)
    acc0 = __builtin_amdgcn_wmma_f32_16x16x32_bf16(
        false, fa0.bf, false, fb.bf, (short)0, acc0, false, false);
    acc1 = __builtin_amdgcn_wmma_f32_16x16x32_bf16(
        false, fa1.bf, false, fb.bf, (short)0, acc1, false, false);
  }
}

#define GSTRIDE(i, n) for (size_t i = blockIdx.x * (size_t)blockDim.x + threadIdx.x; \
                           i < (size_t)(n); i += (size_t)gridDim.x * blockDim.x)

// ---------------- weight prep: convert to bf16 in WMMA-B fragment order ----
// dst[((ntile*KB + kb)*32 + lane)*16 + i] = B[k][n], n = ntile*16 + lane%16,
// k = kb*32 + hb + (i<8 ? i : i+8), hb = (lane/16)*8.
// mode 0: conv  B[k][n] = conv_w[n][k&255][k>>8]      (K = ksz*256, N = 256)
// mode 1: lstm  B[k][n] = k<K0 ? w_ih[n][k] : w_hh[n][k-K0]
// mode 2: out   B[k][n] = out_w[n][k]                 (K = 512, N = 32000)
__global__ void k_swz(int mode, int ksz, int K, int N, int K0,
                      const float* __restrict__ srcA, const float* __restrict__ srcB,
                      unsigned short* __restrict__ dst) {
  int KB = K >> 5;
  size_t total = (size_t)K * N;
  GSTRIDE(idx, total) {
    int i    = (int)(idx & 15);
    int lane = (int)((idx >> 4) & 31);
    size_t rest = idx >> 9;
    int kb    = (int)(rest % KB);
    int ntile = (int)(rest / KB);
    int n  = ntile * 16 + (lane & 15);
    int hb = (lane >> 4) << 3;
    int k  = kb * 32 + hb + (i < 8 ? i : i + 8);
    float v;
    if (mode == 0)      { int j = k >> 8, e = k & 255; v = srcA[((size_t)n * 256 + e) * ksz + j]; }
    else if (mode == 1) { v = (k < K0) ? srcA[(size_t)n * K0 + k]
                                       : srcB[(size_t)n * (K - K0) + (k - K0)]; }
    else                { v = srcA[(size_t)n * K + k]; }
    dst[idx] = f2bf(v);
  }
}

// ---------------- init: zero pooled, d_out[:,0,:], LSTM state ----------------
__global__ void k_init(float* __restrict__ pooled, float* __restrict__ dout,
                       unsigned short* __restrict__ a1, unsigned short* __restrict__ a2,
                       float* __restrict__ c1, float* __restrict__ c2) {
  GSTRIDE(idx, 32 * 32000) {
    size_t b = idx / 32000, n = idx % 32000;
    dout[(b * 64) * 32000 + n] = 0.0f;                    // t = 0 slice
    if (idx < 32 * 768) pooled[idx] = 0.0f;
    if (idx < 32 * 512) {
      size_t bb = idx >> 9, j = idx & 511;
      a1[bb * 1280 + 768 + j] = 0;                        // h1 slot of A1
      c1[idx] = 0.0f; c2[idx] = 0.0f;
    }
    if (idx < 32 * 1024) a2[idx] = 0;                     // [h1,h2] of A2
  }
}

// ---------------- encoder: embedding gather -> bf16 [b][t][e] ----------------
__global__ void k_embed(const int* __restrict__ src, const float* __restrict__ emb,
                        unsigned short* __restrict__ x) {
  GSTRIDE(idx, (size_t)32 * 512 * 256) {
    size_t bt = idx >> 8; int e = (int)(idx & 255);
    x[idx] = f2bf(emb[(size_t)src[bt] * 256 + e]);
  }
}

// ---------------- conv-as-GEMM + bias + relu + maxpool (atomicMax) ----------
// Each wave: two adjacent M-tiles (one tn), sharing B fragments.
__global__ void k_conv(const unsigned short* __restrict__ xemb,
                       const unsigned short* __restrict__ Bswz,
                       const float* __restrict__ bias,
                       float* __restrict__ pooled, int ksz, int foff) {
  int Lk = 513 - ksz;                    // 510/509/508 -> tilesM even
  int K = ksz << 8, KB = K >> 5;
  int nTiles = (((32 * Lk) >> 4) >> 1) * 16;   // (tilesM/2) * 16 col-tiles
  int lane = threadIdx.x & 31;
  int hb = (lane >> 4) << 3;
  int gw = (int)((blockIdx.x * blockDim.x + threadIdx.x) >> 5);
  int nw = (int)((gridDim.x * blockDim.x) >> 5);
  for (int tile = gw; tile < nTiles; tile += nw) {
    int tm0 = (tile >> 4) * 2, tn = tile & 15;
    int m0 = tm0 * 16 + (lane & 15);
    int m1 = m0 + 16;
    int b0 = m0 / Lk, t0 = m0 - b0 * Lk;
    int b1 = m1 / Lk, t1 = m1 - b1 * Lk;
    const unsigned short* a0 = xemb + ((size_t)(b0 * 512 + t0) << 8);
    const unsigned short* a1 = xemb + ((size_t)(b1 * 512 + t1) << 8);
    const unsigned short* blane = Bswz + (size_t)tn * KB * 512 + (size_t)lane * 16;
    v8f acc0 = {}, acc1 = {};
    wmma_kloop2(a0, a1, blane, KB, hb, acc0, acc1);
    int n = tn * 16 + (lane & 15);
    float bv = bias[n];
    int rb = tm0 * 16 + hb;
#pragma unroll
    for (int v = 0; v < 8; ++v) {
      int bb0 = (rb + v) / Lk;
      float v0 = fmaxf(acc0[v] + bv, 0.0f);
      atomicMax((int*)(pooled + (size_t)bb0 * 768 + foff + n), __float_as_int(v0));
      int bb1 = (rb + 16 + v) / Lk;
      float v1 = fmaxf(acc1[v] + bv, 0.0f);
      atomicMax((int*)(pooled + (size_t)bb1 * 768 + foff + n), __float_as_int(v1));
    }
  }
}

// ---------------- M=32 GEMM (LSTM gates): C[32][N] = A[32][K] * Bswz --------
// Both M-tiles per wave (M is exactly 32), B fragment loaded once.
__global__ void k_gemm32(const unsigned short* __restrict__ A, int lda,
                         const unsigned short* __restrict__ Bswz, int K, int N,
                         float* __restrict__ C) {
  int KB = K >> 5, tilesN = N >> 4;
  int lane = threadIdx.x & 31;
  int hb = (lane >> 4) << 3;
  int gw = (int)((blockIdx.x * blockDim.x + threadIdx.x) >> 5);
  int nw = (int)((gridDim.x * blockDim.x) >> 5);
  for (int tn = gw; tn < tilesN; tn += nw) {
    const unsigned short* a0 = A + (size_t)(lane & 15) * lda;
    const unsigned short* a1 = A + (size_t)(16 + (lane & 15)) * lda;
    const unsigned short* blane = Bswz + (size_t)tn * KB * 512 + (size_t)lane * 16;
    v8f acc0 = {}, acc1 = {};
    wmma_kloop2(a0, a1, blane, KB, hb, acc0, acc1);
    int n = tn * 16 + (lane & 15);
#pragma unroll
    for (int v = 0; v < 8; ++v) {
      C[(size_t)(hb + v) * N + n]      = acc0[v];
      C[(size_t)(16 + hb + v) * N + n] = acc1[v];
    }
  }
}

// ---------------- logits: (32x512 bf16 h2) @ Wout_swz + out_b -> d_out ------
__global__ void k_logits(const unsigned short* __restrict__ A2,
                         const unsigned short* __restrict__ Bswz,
                         const float* __restrict__ outb,
                         float* __restrict__ dout, int tpos) {
  const int KB = 16, tilesN = 2000;                       // K=512, N=32000
  int lane = threadIdx.x & 31;
  int hb = (lane >> 4) << 3;
  int gw = (int)((blockIdx.x * blockDim.x + threadIdx.x) >> 5);
  int nw = (int)((gridDim.x * blockDim.x) >> 5);
  for (int tn = gw; tn < tilesN; tn += nw) {
    const unsigned short* a0 = A2 + (size_t)(lane & 15) * 1024 + 512;        // h2 half
    const unsigned short* a1 = A2 + (size_t)(16 + (lane & 15)) * 1024 + 512;
    const unsigned short* blane = Bswz + (size_t)tn * KB * 512 + (size_t)lane * 16;
    v8f acc0 = {}, acc1 = {};
    wmma_kloop2(a0, a1, blane, KB, hb, acc0, acc1);
    int n = tn * 16 + (lane & 15);
    float bv = outb[n];
#pragma unroll
    for (int v = 0; v < 8; ++v) {
      dout[((size_t)(hb + v) * 64 + tpos) * 32000 + n]      = acc0[v] + bv;
      dout[((size_t)(16 + hb + v) * 64 + tpos) * 32000 + n] = acc1[v] + bv;
    }
  }
}

// ---------------- small f32 FC (negligible FLOPs; keep full precision) ------
__global__ void k_fc(const float* __restrict__ in, const float* __restrict__ w,
                     const float* __restrict__ bias, float* __restrict__ out,
                     unsigned short* __restrict__ outbf, int K, int relu) {
  size_t idx = blockIdx.x * (size_t)blockDim.x + threadIdx.x;
  if (idx >= 32 * 512) return;
  int b = (int)(idx >> 9), h = (int)(idx & 511);
  const float* row = w + (size_t)h * K;
  const float* x   = in + (size_t)b * K;
  float s = bias[h];
  for (int k = 0; k < K; ++k) s += x[k] * row[k];
  if (relu) s = fmaxf(s, 0.0f);
  out[idx] = s;
  if (outbf) outbf[idx] = f2bf(s);
}

// ---------------- decoder inputs X[t][b][0:768] = [dec_emb[tok], enc_out] ---
__global__ void k_prep_dec(const int* __restrict__ trg, const float* __restrict__ dec_emb,
                           const unsigned short* __restrict__ encbf,
                           unsigned short* __restrict__ X) {
  GSTRIDE(idx, (size_t)63 * 32 * 768) {
    int j = (int)(idx % 768);
    size_t rest = idx / 768;
    int b = (int)(rest & 31), t = (int)(rest >> 5);
    unsigned short v;
    if (j < 256) v = f2bf(dec_emb[(size_t)trg[b * 64 + t] * 256 + j]);
    else         v = encbf[b * 512 + (j - 256)];
    X[idx] = v;
  }
}

__global__ void k_build_a1(const unsigned short* __restrict__ X,
                           unsigned short* __restrict__ A1, int t) {
  GSTRIDE(idx, 32 * 768) {
    size_t b = idx / 768, j = idx % 768;
    A1[b * 1280 + j] = X[((size_t)t * 32 + b) * 768 + j];
  }
}

// ---------------- LSTM pointwise cell (f32) --------------------------------
__global__ void k_cell(const float* __restrict__ gates, const float* __restrict__ bih,
                       const float* __restrict__ bhh, float* __restrict__ c,
                       unsigned short* __restrict__ h1dst, int s1, int o1,
                       unsigned short* __restrict__ h2dst, int s2, int o2) {
  size_t idx = blockIdx.x * (size_t)blockDim.x + threadIdx.x;
  if (idx >= 32 * 512) return;
  int b = (int)(idx >> 9), j = (int)(idx & 511);
  const float* g = gates + (size_t)b * 2048;
  float gi = g[j]        + bih[j]        + bhh[j];
  float gf = g[512  + j] + bih[512  + j] + bhh[512  + j];
  float gg = g[1024 + j] + bih[1024 + j] + bhh[1024 + j];
  float go = g[1536 + j] + bih[1536 + j] + bhh[1536 + j];
  float si = 1.0f / (1.0f + __expf(-gi));
  float sf = 1.0f / (1.0f + __expf(-gf));
  float so = 1.0f / (1.0f + __expf(-go));
  float cn = sf * c[idx] + si * tanhf(gg);
  c[idx] = cn;
  unsigned short hb_ = f2bf(so * tanhf(cn));
  h1dst[(size_t)b * s1 + o1 + j] = hb_;
  if (h2dst) h2dst[(size_t)b * s2 + o2 + j] = hb_;
}

// ---------------- host orchestration ----------------------------------------
extern "C" void kernel_launch(void* const* d_in, const int* in_sizes, int n_in,
                              void* d_out, int out_size, void* d_ws, size_t ws_size,
                              hipStream_t stream) {
  const int*   src     = (const int*)d_in[0];
  const int*   trg     = (const int*)d_in[1];
  const float* enc_emb = (const float*)d_in[2];
  const float* dec_emb = (const float*)d_in[3];
  const float* cw3 = (const float*)d_in[4];  const float* cb3 = (const float*)d_in[5];
  const float* cw4 = (const float*)d_in[6];  const float* cb4 = (const float*)d_in[7];
  const float* cw5 = (const float*)d_in[8];  const float* cb5 = (const float*)d_in[9];
  const float* fc1w = (const float*)d_in[10]; const float* fc1b = (const float*)d_in[11];
  const float* fc2w = (const float*)d_in[12]; const float* fc2b = (const float*)d_in[13];
  const float* wih1 = (const float*)d_in[14]; const float* whh1 = (const float*)d_in[15];
  const float* bih1 = (const float*)d_in[16]; const float* bhh1 = (const float*)d_in[17];
  const float* wih2 = (const float*)d_in[18]; const float* whh2 = (const float*)d_in[19];
  const float* bih2 = (const float*)d_in[20]; const float* bhh2 = (const float*)d_in[21];
  const float* outw = (const float*)d_in[22]; const float* outb = (const float*)d_in[23];
  float* out = (float*)d_out;

  char* ws = (char*)d_ws;
  size_t off = 0;
  auto alloc = [&](size_t bytes) -> void* {
    void* p = ws + off;
    off = (off + bytes + 255) & ~(size_t)255;
    return p;
  };
  unsigned short* Wout = (unsigned short*)alloc((size_t)512 * 32000 * 2);  // 32.8 MB, L2-resident
  unsigned short* Wc1  = (unsigned short*)alloc((size_t)1280 * 2048 * 2);
  unsigned short* Wc2  = (unsigned short*)alloc((size_t)1024 * 2048 * 2);
  unsigned short* Wg3  = (unsigned short*)alloc((size_t)768  * 256 * 2);
  unsigned short* Wg4  = (unsigned short*)alloc((size_t)1024 * 256 * 2);
  unsigned short* Wg5  = (unsigned short*)alloc((size_t)1280 * 256 * 2);
  unsigned short* Xemb = (unsigned short*)alloc((size_t)32 * 512 * 256 * 2);
  unsigned short* Xdec = (unsigned short*)alloc((size_t)63 * 32 * 768 * 2);
  float* pooled = (float*)alloc(32 * 768 * 4);
  float* hfc    = (float*)alloc(32 * 512 * 4);
  float* encf   = (float*)alloc(32 * 512 * 4);
  unsigned short* encbf = (unsigned short*)alloc(32 * 512 * 2);
  unsigned short* A1 = (unsigned short*)alloc(32 * 1280 * 2);   // [x_t | h1] bf16
  unsigned short* A2 = (unsigned short*)alloc(32 * 1024 * 2);   // [h1 | h2] bf16
  float* g1 = (float*)alloc(32 * 2048 * 4);
  float* g2 = (float*)alloc(32 * 2048 * 4);
  float* c1 = (float*)alloc(32 * 512 * 4);
  float* c2 = (float*)alloc(32 * 512 * 4);

  dim3 B(256);
  // init + one-time bf16 weight prep (captured in graph; deterministic)
  k_init<<<1024, B, 0, stream>>>(pooled, out, A1, A2, c1, c2);
  k_swz<<<2048, B, 0, stream>>>(2, 0, 512, 32000, 0, outw, nullptr, Wout);
  k_swz<<<512,  B, 0, stream>>>(1, 0, 1280, 2048, 768, wih1, whh1, Wc1);
  k_swz<<<512,  B, 0, stream>>>(1, 0, 1024, 2048, 512, wih2, whh2, Wc2);
  k_swz<<<128,  B, 0, stream>>>(0, 3, 768,  256, 0, cw3, nullptr, Wg3);
  k_swz<<<128,  B, 0, stream>>>(0, 4, 1024, 256, 0, cw4, nullptr, Wg4);
  k_swz<<<160,  B, 0, stream>>>(0, 5, 1280, 256, 0, cw5, nullptr, Wg5);
  k_embed<<<2048, B, 0, stream>>>(src, enc_emb, Xemb);
  // encoder convs (WMMA GEMM + fused relu/maxpool, 2 M-tiles per wave)
  k_conv<<<1024, B, 0, stream>>>(Xemb, Wg3, cb3, pooled, 3, 0);
  k_conv<<<1024, B, 0, stream>>>(Xemb, Wg4, cb4, pooled, 4, 256);
  k_conv<<<1024, B, 0, stream>>>(Xemb, Wg5, cb5, pooled, 5, 512);
  // encoder FCs
  k_fc<<<64, B, 0, stream>>>(pooled, fc1w, fc1b, hfc, nullptr, 768, 1);
  k_fc<<<64, B, 0, stream>>>(hfc, fc2w, fc2b, encf, encbf, 512, 0);
  // decoder inputs for all steps
  k_prep_dec<<<1024, B, 0, stream>>>(trg, dec_emb, encbf, Xdec);
  // sequential decoder
  for (int t = 0; t < 63; ++t) {
    k_build_a1<<<96, B, 0, stream>>>(Xdec, A1, t);
    k_gemm32<<<16, B, 0, stream>>>(A1, 1280, Wc1, 1280, 2048, g1);
    k_cell<<<64, B, 0, stream>>>(g1, bih1, bhh1, c1, A1, 1280, 768, A2, 1024, 0);
    k_gemm32<<<16, B, 0, stream>>>(A2, 1024, Wc2, 1024, 2048, g2);
    k_cell<<<64, B, 0, stream>>>(g2, bih2, bhh2, c2, A2, 1024, 512, nullptr, 0, 0);
    k_logits<<<250, B, 0, stream>>>(A2, Wout, outb, out, t + 1);
  }
}
